// Attention_20418274525612
// MI455X (gfx1250) — matrix operand order repaired and checked by reading, compiled, not run on hardware
//
#include <hip/hip_runtime.h>

typedef _Float16 v16h __attribute__((ext_vector_type(16)));
typedef _Float16 v8h  __attribute__((ext_vector_type(8)));
typedef float    v8f  __attribute__((ext_vector_type(8)));
typedef float    f32x4 __attribute__((ext_vector_type(4)));
typedef unsigned int u32x4 __attribute__((ext_vector_type(4)));
typedef int      i32x4 __attribute__((ext_vector_type(4)));
typedef int      i32x8 __attribute__((ext_vector_type(8)));

#define DEVINL __device__ __forceinline__

constexpr int Bb   = 2;
constexpr int Nseq = 2048;
constexpr int Cin  = 256;
constexpr int Hh   = 8;
constexpr int Ch   = 32;

// Tensor-Data-Mover availability (device pass only; host pass sees fallback).
#if defined(__HIP_DEVICE_COMPILE__) &&                                         \
    __has_builtin(__builtin_amdgcn_tensor_load_to_lds) &&                      \
    __has_builtin(__builtin_amdgcn_s_wait_tensorcnt)
#define USE_TDM 1
#else
#define USE_TDM 0
#endif

// Attention-kernel dynamic LDS layout (fixed regardless of TDM path so the
// host-side launch size always matches):
//   [0, 16KB)  : bias tiles, 4 waves x 2 buffers x (16 rows x 32 f32)
//   [16KB, +2K): transposed V tile Vt[c][k] (f16)
constexpr int BIAS_BUF  = 16 * 32 * 4;                 // 2 KB
constexpr int VT_OFF    = 4 * 2 * BIAS_BUF;            // 16 KB
constexpr int SMEM_ATTN = VT_OFF + 32 * 32 * 2;        // 18 KB

DEVINL v8f wmma16x16x32(v16h a, v16h b, v8f c) {
  return __builtin_amdgcn_wmma_f32_16x16x32_f16(false, a, false, b, (short)0, c,
                                                false, false);
}

// A-fragment (16x32 f16) from row-major storage (`stride` in halfs).
// lanes 0-15 row=lane, K 0..15; lanes 16-31 row=lane-16, K 16..31.
DEVINL v16h load_fragA(const _Float16* base, int lane, int stride) {
  const int row = lane & 15;
  const int kb  = (lane >> 4) << 4;
  const _Float16* p = base + row * stride + kb;
  v8h lo = *(const v8h*)(p);
  v8h hi = *(const v8h*)(p + 8);
  return __builtin_shufflevector(lo, hi, 0, 1, 2, 3, 4, 5, 6, 7,
                                 8, 9, 10, 11, 12, 13, 14, 15);
}

// B-fragment (32x16 f16) from column-contiguous storage (B[k][n] at base[n*stride+k]).
// Contiguous k, as in load_fragA: lanes 0-15 col=lane hold K=0..15; lanes 16-31 col=lane-16 hold K=16..31.
DEVINL v16h load_fragB(const _Float16* base, int lane, int stride) {
  const int col = lane & 15;
  const int kb  = (lane >> 4) << 4;
  const _Float16* p = base + col * stride + kb;
  v8h lo = *(const v8h*)(p);
  v8h hi = *(const v8h*)(p + 8);
  return __builtin_shufflevector(lo, hi, 0, 1, 2, 3, 4, 5, 6, 7,
                                 8, 9, 10, 11, 12, 13, 14, 15);
}

#if USE_TDM
// Issue a TDM load of a 16(row) x 32(f32) bias tile into LDS at `lds_off`.
// D# built per CDNA5 ISA ch.8: group0 {count=1, lds_addr, global_addr, type=2},
// group1 {mask=0, data_size=4B, tensor_dim0/1=2048, tile=[32,16], stride0=2048}.
DEVINL void tdm_load_bias(unsigned lds_off, const float* gsrc) {
  unsigned long long ga = (unsigned long long)gsrc;
  u32x4 g0;
  g0[0] = 1u;                                            // count=1, user D#
  g0[1] = lds_off;                                       // LDS byte address
  g0[2] = (unsigned)ga;                                  // global_addr[31:0]
  g0[3] = (unsigned)((ga >> 32) & 0x01FFFFFFull)         // global_addr[56:32]
          | (2u << 30);                                  // type = 2 ("image")
  i32x8 g1;
  g1[0] = (int)(2u << 16);                               // data_size=4B, mask=0
  g1[1] = (int)(((unsigned)Nseq & 0xFFFFu) << 16);       // tensor_dim0 lo16
  g1[2] = (int)(((unsigned)Nseq & 0xFFFFu) << 16);       // dim0 hi=0 | dim1 lo16
  g1[3] = (int)(32u << 16);                              // dim1 hi=0 | tile_dim0=32
  g1[4] = 16;                                            // tile_dim1=16, tile_dim2=0
  g1[5] = Nseq;                                          // tensor_dim0_stride lo32
  g1[6] = 0;                                             // stride hi | dim1_stride lo
  g1[7] = 0;
  i32x4 z4 = {0, 0, 0, 0};
#if __clang_major__ >= 23
  i32x8 z8 = {0, 0, 0, 0, 0, 0, 0, 0};
  __builtin_amdgcn_tensor_load_to_lds(g0, g1, z4, z4, z8, 0);
#else
  __builtin_amdgcn_tensor_load_to_lds(g0, g1, z4, z4, 0);
#endif
}
#endif

// ---------------------------------------------------------------------------
// Kernel 1: projections, templated on Z = 0:Q 1:K 2:V 3:gate.
// Block 256 thr (8 waves), tile 128(M) x 64(N), K-loop by 32.
// ---------------------------------------------------------------------------
template <int Z>
__global__ __launch_bounds__(256) void proj_kernel(
    const float* __restrict__ X, const float* __restrict__ W,
    const float* __restrict__ bg,
    _Float16* __restrict__ Hout, float* __restrict__ G) {
  __shared__ _Float16 As[128 * 32];   // activation tile [m][k]
  __shared__ _Float16 Bs[64 * 32];    // weight tile transposed [n][k]

  const int m0 = blockIdx.x * 128;
  const int n0 = blockIdx.y * 64;
  const int t = threadIdx.x, lane = t & 31, wave = t >> 5;
  const int mw = (wave >> 1) * 32, nw = (wave & 1) * 32;

  v8f acc[2][2] = {};

  for (int kk = 0; kk < Cin; kk += 32) {
    __syncthreads();
    {  // stage A: 128x32 f32 -> f16
      const int row = t >> 1, c0 = (t & 1) * 16;
      const float* src = X + (size_t)(m0 + row) * Cin + kk + c0;
      _Float16* dst = As + row * 32 + c0;
#pragma unroll
      for (int i = 0; i < 16; i += 4) {
        f32x4 f = *(const f32x4*)(src + i);
        dst[i + 0] = (_Float16)f[0];
        dst[i + 1] = (_Float16)f[1];
        dst[i + 2] = (_Float16)f[2];
        dst[i + 3] = (_Float16)f[3];
      }
    }
    {  // stage B transposed: W[kk+r][n0+c] -> Bs[c][r]
      const int r = t >> 3, c0 = (t & 7) * 8;
      const float* src = W + (size_t)(kk + r) * Cin + n0 + c0;
#pragma unroll
      for (int i = 0; i < 8; i += 4) {
        f32x4 f = *(const f32x4*)(src + i);
        Bs[(c0 + i + 0) * 32 + r] = (_Float16)f[0];
        Bs[(c0 + i + 1) * 32 + r] = (_Float16)f[1];
        Bs[(c0 + i + 2) * 32 + r] = (_Float16)f[2];
        Bs[(c0 + i + 3) * 32 + r] = (_Float16)f[3];
      }
    }
    __syncthreads();
    v16h a0 = load_fragA(As + (mw + 0) * 32, lane, 32);
    v16h a1 = load_fragA(As + (mw + 16) * 32, lane, 32);
    v16h b0 = load_fragB(Bs + (nw + 0) * 32, lane, 32);
    v16h b1 = load_fragB(Bs + (nw + 16) * 32, lane, 32);
    acc[0][0] = wmma16x16x32(a0, b0, acc[0][0]);
    acc[0][1] = wmma16x16x32(a0, b1, acc[0][1]);
    acc[1][0] = wmma16x16x32(a1, b0, acc[1][0]);
    acc[1][1] = wmma16x16x32(a1, b1, acc[1][1]);
  }

  const int colb = lane & 15;
  const int rowoff = (lane >> 4) * 8;
  const float qscale = 0.17677669529663687f;  // 1/sqrt(32)
#pragma unroll
  for (int i = 0; i < 2; ++i)
#pragma unroll
    for (int j = 0; j < 2; ++j)
#pragma unroll
      for (int r = 0; r < 8; ++r) {
        const int gm = m0 + mw + i * 16 + rowoff + r;
        const int co = n0 + nw + j * 16 + colb;
        const float v = acc[i][j][r];
        if (Z == 3) {
          const float s = v + bg[co];
          G[(size_t)gm * Cin + co] = 1.0f / (1.0f + __expf(-s));
        } else {
          const int b = gm >> 11, nidx = gm & (Nseq - 1);
          const int h = co >> 5, c = co & 31;
          const size_t hidx = (((size_t)(b * Hh + h)) * Nseq + nidx) * Ch + c;
          Hout[hidx] = (_Float16)(Z == 0 ? v * qscale : v);
        }
      }
}

// ---------------------------------------------------------------------------
// Kernel 2: flash attention with pairwise bias + sigmoid gating.
// Block 128 thr (4 waves); each wave owns 16 query rows; 32 keys per step.
// S^T = K_tile @ Q^T + bias  (bias tile fed directly as the WMMA C operand:
// the TDM-streamed tile is already in C/D register layout), online softmax
// via xor-16 shuffles, P remapped (one xor-16 swap) into the PV B-fragment,
// O^T = V^T @ P^T in f32.  Bias tiles stream through the Tensor Data Mover
// into a per-wave LDS double buffer (TENSORcnt-pipelined); fallback =
// nontemporal global loads.
// ---------------------------------------------------------------------------
__global__ __launch_bounds__(128) void attn_kernel(
    const _Float16* __restrict__ Qh, const _Float16* __restrict__ Kh,
    const _Float16* __restrict__ Vh, const float* __restrict__ bias,
    const float* __restrict__ G, _Float16* __restrict__ Og) {
  extern __shared__ char smem[];
  _Float16* Vt = (_Float16*)(smem + VT_OFF);   // Vt[c][k]

  const int bh = blockIdx.y;                   // b*H + h
  const int t = threadIdx.x, lane = t & 31;
  const int wave = __builtin_amdgcn_readfirstlane(t >> 5);
  const int qtile = blockIdx.x * 64 + wave * 16;

  const size_t base = (size_t)bh * Nseq * Ch;
  const _Float16* Qb = Qh + base;
  const _Float16* Kb = Kh + base;
  const _Float16* Vb = Vh + base;
  const float* biasq = bias + (size_t)bh * Nseq * Nseq + (size_t)qtile * Nseq;

  const v16h qfrag = load_fragB(Qb + (size_t)qtile * Ch, lane, Ch);
  const int q = qtile + (lane & 15);
  const int koff = (lane >> 4) * 8;            // 0 or 8
  const bool lolane = (lane < 16);

  v8f accA = {}, accB = {};                    // O^T channels 0..15 / 16..31
  float mrow = -3.0e38f, lrow = 0.0f;

  constexpr int NT = Nseq / 32;
#if USE_TDM
  tdm_load_bias((unsigned)(wave * 2 * BIAS_BUF), biasq);   // prime buffer 0
#endif

  for (int j = 0; j < NT; ++j) {
#if USE_TDM
    if (j + 1 < NT)  // async prefetch of next bias tile into the other buffer
      tdm_load_bias((unsigned)((wave * 2 + ((j + 1) & 1)) * BIAS_BUF),
                    biasq + (size_t)(j + 1) * 32);
#endif
    __syncthreads();
    {  // cooperatively stage V (32 keys x 32 ch) transposed into LDS
      const int key = t >> 2, c0 = (t & 3) * 8;
      v8h v = *(const v8h*)(Vb + (size_t)(j * 32 + key) * Ch + c0);
#pragma unroll
      for (int i = 0; i < 8; ++i) Vt[(c0 + i) * 32 + key] = v[i];
    }
    __syncthreads();

    // pairwise bias tile — already in the WMMA C/D register layout
    // (lane = query column, register r = key row)
    f32x4 f0, f1, f2, f3;
#if USE_TDM
    if (j + 1 < NT) __builtin_amdgcn_s_wait_tensorcnt(1);
    else            __builtin_amdgcn_s_wait_tensorcnt(0);
    const float* bp = (const float*)(smem + (wave * 2 + (j & 1)) * BIAS_BUF) +
                      (lane & 15) * 32 + koff;
    f0 = *(const f32x4*)(bp + 0);
    f1 = *(const f32x4*)(bp + 4);
    f2 = *(const f32x4*)(bp + 16);
    f3 = *(const f32x4*)(bp + 20);
#else
    const float* bp = biasq + (size_t)(lane & 15) * Nseq + j * 32 + koff;
    f0 = __builtin_nontemporal_load((const f32x4*)(bp + 0));
    f1 = __builtin_nontemporal_load((const f32x4*)(bp + 4));
    f2 = __builtin_nontemporal_load((const f32x4*)(bp + 16));
    f3 = __builtin_nontemporal_load((const f32x4*)(bp + 20));
#endif
    v8f clo = {f0[0], f0[1], f0[2], f0[3], f1[0], f1[1], f1[2], f1[3]};
    v8f chi = {f2[0], f2[1], f2[2], f2[3], f3[0], f3[1], f3[2], f3[3]};

    // S^T tiles: rows = keys, cols = queries;  D = K @ Q^T + bias
    v16h klo = load_fragA(Kb + (size_t)(j * 32 + 0) * Ch, lane, Ch);
    v16h khi = load_fragA(Kb + (size_t)(j * 32 + 16) * Ch, lane, Ch);
    v8f slo = wmma16x16x32(klo, qfrag, clo);   // keys j*32 + {0..15}
    v8f shi = wmma16x16x32(khi, qfrag, chi);   // keys j*32 + {16..31}

    // online softmax (lanes l and l+16 hold disjoint key subsets of column q)
    float tm = slo[0];
#pragma unroll
    for (int r = 1; r < 8; ++r) tm = fmaxf(tm, slo[r]);
#pragma unroll
    for (int r = 0; r < 8; ++r) tm = fmaxf(tm, shi[r]);
    tm = fmaxf(tm, __shfl_xor(tm, 16, 32));
    const float mnew = fmaxf(mrow, tm);
    const float alpha = __expf(mrow - mnew);

    float plo[8], phi[8], ts = 0.0f;
#pragma unroll
    for (int r = 0; r < 8; ++r) { plo[r] = __expf(slo[r] - mnew); ts += plo[r]; }
#pragma unroll
    for (int r = 0; r < 8; ++r) { phi[r] = __expf(shi[r] - mnew); ts += phi[r]; }
    ts += __shfl_xor(ts, 16, 32);
    lrow = lrow * alpha + ts;
    mrow = mnew;

    // C/D layout (M-split over lane halves) -> B-frag layout (K-split):
    // one xor-16 exchange of 8 values.
    v16h pfrag;
#pragma unroll
    for (int r = 0; r < 8; ++r) {
      float xch = __shfl_xor(lolane ? phi[r] : plo[r], 16, 32);
      pfrag[r]     = (_Float16)(lolane ? plo[r] : xch);
      pfrag[8 + r] = (_Float16)(lolane ? xch : phi[r]);
    }

#pragma unroll
    for (int r = 0; r < 8; ++r) { accA[r] *= alpha; accB[r] *= alpha; }

    v16h v0 = load_fragA(Vt, lane, 32);            // channels  0..15
    v16h v1 = load_fragA(Vt + 16 * 32, lane, 32);  // channels 16..31
    accA = wmma16x16x32(v0, pfrag, accA);
    accB = wmma16x16x32(v1, pfrag, accB);
  }

  // normalize, apply sigmoid gate, emit f16 for the output GEMM
  const float invl = 1.0f / lrow;
  const int b = bh >> 3, h = bh & 7;
  const size_t orow = ((size_t)b * Nseq + q) * (Hh * Ch) + h * Ch;
  const float* gp = G + orow;
  _Float16* op = Og + orow;
#pragma unroll
  for (int r = 0; r < 8; ++r) {
    const int cA = koff + r;
    const int cB = 16 + koff + r;
    op[cA] = (_Float16)(accA[r] * invl * gp[cA]);
    op[cB] = (_Float16)(accB[r] * invl * gp[cB]);
  }
}

// ---------------------------------------------------------------------------
// Kernel 3: out = Og(f16) @ Wo + bo  (f32, nontemporal stores)
// ---------------------------------------------------------------------------
__global__ __launch_bounds__(256) void out_kernel(
    const _Float16* __restrict__ Og, const float* __restrict__ Wo,
    const float* __restrict__ bo, float* __restrict__ out) {
  __shared__ _Float16 As[128 * 32];
  __shared__ _Float16 Bs[64 * 32];

  const int m0 = blockIdx.x * 128, n0 = blockIdx.y * 64;
  const int t = threadIdx.x, lane = t & 31, wave = t >> 5;
  const int mw = (wave >> 1) * 32, nw = (wave & 1) * 32;

  v8f acc[2][2] = {};

  for (int kk = 0; kk < Cin; kk += 32) {
    __syncthreads();
    {  // stage A (already f16)
      const int row = t >> 1, c0 = (t & 1) * 16;
      const _Float16* src = Og + (size_t)(m0 + row) * Cin + kk + c0;
      *(v8h*)(As + row * 32 + c0)     = *(const v8h*)(src);
      *(v8h*)(As + row * 32 + c0 + 8) = *(const v8h*)(src + 8);
    }
    {  // stage B transposed, f32 -> f16
      const int r = t >> 3, c0 = (t & 7) * 8;
      const float* src = Wo + (size_t)(kk + r) * Cin + n0 + c0;
#pragma unroll
      for (int i = 0; i < 8; i += 4) {
        f32x4 f = *(const f32x4*)(src + i);
        Bs[(c0 + i + 0) * 32 + r] = (_Float16)f[0];
        Bs[(c0 + i + 1) * 32 + r] = (_Float16)f[1];
        Bs[(c0 + i + 2) * 32 + r] = (_Float16)f[2];
        Bs[(c0 + i + 3) * 32 + r] = (_Float16)f[3];
      }
    }
    __syncthreads();
    v16h a0 = load_fragA(As + (mw + 0) * 32, lane, 32);
    v16h a1 = load_fragA(As + (mw + 16) * 32, lane, 32);
    v16h b0 = load_fragB(Bs + (nw + 0) * 32, lane, 32);
    v16h b1 = load_fragB(Bs + (nw + 16) * 32, lane, 32);
    acc[0][0] = wmma16x16x32(a0, b0, acc[0][0]);
    acc[0][1] = wmma16x16x32(a0, b1, acc[0][1]);
    acc[1][0] = wmma16x16x32(a1, b0, acc[1][0]);
    acc[1][1] = wmma16x16x32(a1, b1, acc[1][1]);
  }

  const int colb = lane & 15;
  const int rowoff = (lane >> 4) * 8;
#pragma unroll
  for (int i = 0; i < 2; ++i)
#pragma unroll
    for (int j = 0; j < 2; ++j)
#pragma unroll
      for (int r = 0; r < 8; ++r) {
        const int gm = m0 + mw + i * 16 + rowoff + r;
        const int co = n0 + nw + j * 16 + colb;
        __builtin_nontemporal_store(acc[i][j][r] + bo[co],
                                    &out[(size_t)gm * Cin + co]);
      }
}

// ---------------------------------------------------------------------------
extern "C" void kernel_launch(void* const* d_in, const int* in_sizes, int n_in,
                              void* d_out, int out_size, void* d_ws,
                              size_t ws_size, hipStream_t stream) {
  const float* qx   = (const float*)d_in[0];
  const float* kvx  = (const float*)d_in[1];
  const float* bias = (const float*)d_in[2];
  const float* Wq   = (const float*)d_in[3];
  const float* Wk   = (const float*)d_in[4];
  const float* Wv   = (const float*)d_in[5];
  const float* Wg   = (const float*)d_in[6];
  const float* bg   = (const float*)d_in[7];
  const float* Wo   = (const float*)d_in[8];
  const float* bo   = (const float*)d_in[9];
  float* out = (float*)d_out;

  constexpr size_t ELT = (size_t)Bb * Hh * Nseq * Ch;  // 1<<20
  char* ws = (char*)d_ws;
  _Float16* Qh = (_Float16*)ws;                 // 2 MB
  _Float16* Kh = Qh + ELT;                      // 2 MB
  _Float16* Vh = Kh + ELT;                      // 2 MB
  float*    G  = (float*)(Vh + ELT);            // 4 MB
  _Float16* Og = (_Float16*)(G + ELT);          // 2 MB

  dim3 g1((Bb * Nseq) / 128, Cin / 64);
  proj_kernel<0><<<g1, 256, 0, stream>>>(qx,  Wq, bg, Qh, G);
  proj_kernel<1><<<g1, 256, 0, stream>>>(kvx, Wk, bg, Kh, G);
  proj_kernel<2><<<g1, 256, 0, stream>>>(kvx, Wv, bg, Vh, G);
  proj_kernel<3><<<g1, 256, 0, stream>>>(qx,  Wg, bg, Qh, G);

  dim3 g2(Nseq / 64, Bb * Hh);
  attn_kernel<<<g2, 128, SMEM_ATTN, stream>>>(Qh, Kh, Vh, bias, G, Og);

  dim3 g3((Bb * Nseq) / 128, Cin / 64);
  out_kernel<<<g3, 256, 0, stream>>>(Og, Wo, bo, out);
}
